// DF11Linear_50422916055141
// MI455X (gfx1250) — matrix operand order repaired and checked by reading, compiled, not run on hardware
//
#include <hip/hip_runtime.h>

#define IN_F 4096
#define OUT_F 4096
#define BATCH 16
#define WAVES_PER_BLOCK 16
#define K_CHUNK (IN_F / WAVES_PER_BLOCK)   // 256 per wave -> 8 WMMA steps

typedef __attribute__((ext_vector_type(16))) __bf16        v16bf;
typedef __attribute__((ext_vector_type(2)))  __bf16        v2bf;
typedef __attribute__((ext_vector_type(8)))  float         v8f;
typedef __attribute__((ext_vector_type(8)))  unsigned int  v8u;

// f32 pair -> packed bf16 dword, round-to-nearest-even (matches jnp astype(bfloat16))
__device__ __forceinline__ unsigned pack_bf16_rne(float lo, float hi) {
#if __has_builtin(__builtin_amdgcn_cvt_pk_bf16_f32)
    v2bf p = __builtin_amdgcn_cvt_pk_bf16_f32(lo, hi);
    return __builtin_bit_cast(unsigned, p);
#else
    unsigned ul = __builtin_bit_cast(unsigned, lo);
    unsigned uh = __builtin_bit_cast(unsigned, hi);
    ul = (ul + 0x7fffu + ((ul >> 16) & 1u)) >> 16;
    uh = (uh + 0x7fffu + ((uh >> 16) & 1u)) >> 16;
    return ul | (uh << 16);
#endif
}

// One 16x16x32 bf16 WMMA step: build A from x (f32->bf16), decode B from DF11
// metadata, accumulate into c.
__device__ __forceinline__ v8f df11_step(int k0, v8f c,
                                         const float* __restrict__ xrow,
                                         const int*   __restrict__ exp_idx,
                                         const int*   __restrict__ sign_mant,
                                         const unsigned* ldsLutSh,  // 2x replicated, lut<<7
                                         unsigned lutBase,          // (lane&1)<<5
                                         int lane, int laneHalf, int n0) {
    // ---- A operand: x[16, k0:k0+32] as bf16 (16-bit A 16x32 layout) ----
    v8u au;
#pragma unroll
    for (int v = 0; v < 8; ++v) {
        const int kb = k0 + 2 * v + ((v >= 4) ? 8 : 0) + 8 * laneHalf;
        au[v] = pack_bf16_rne(xrow[kb], xrow[kb + 1]);
    }

    // ---- B operand: decode W[n0:n0+16, k0:k0+32] -> bf16, B is K x N ----
    // lane <-> N = lane&15; VGPR v halves <-> K pair (same order as A)
    const size_t col = (size_t)(n0 + (lane & 15)) * IN_F;
    v8u bu;
#pragma unroll
    for (int v = 0; v < 8; ++v) {
        const size_t r0 = col + (size_t)(k0 + 2 * v + ((v >= 4) ? 8 : 0) + 8 * laneHalf);  // K even
        const size_t r1 = r0 + 1;                             // K odd
        const unsigned e0 = (unsigned)__builtin_nontemporal_load(exp_idx   + r0);
        const unsigned s0 = (unsigned)__builtin_nontemporal_load(sign_mant + r0);
        const unsigned e1 = (unsigned)__builtin_nontemporal_load(exp_idx   + r1);
        const unsigned s1 = (unsigned)__builtin_nontemporal_load(sign_mant + r1);
        // bf16 bits: sign(1) | exp(8) | mant(7); LUT stored as exp<<7, 2x replicated
        const unsigned w0 = ((s0 >> 7) << 15) | ldsLutSh[lutBase | (e0 & 31)] | (s0 & 0x7fu);
        const unsigned w1 = ((s1 >> 7) << 15) | ldsLutSh[lutBase | (e1 & 31)] | (s1 & 0x7fu);
        bu[v] = (w0 & 0xffffu) | (w1 << 16);
    }

    const v16bf a = __builtin_bit_cast(v16bf, au);
    const v16bf b = __builtin_bit_cast(v16bf, bu);
    return __builtin_amdgcn_wmma_f32_16x16x32_bf16(false, a, false, b,
                                                   (short)0, c, false, false);
}

__global__ __launch_bounds__(WAVES_PER_BLOCK * 32)
void df11_linear_wmma_kernel(const float* __restrict__ x,
                             const int*   __restrict__ exp_idx,
                             const int*   __restrict__ sign_mant,
                             const int*   __restrict__ lut_exp,
                             const float* __restrict__ bias,
                             float*       __restrict__ out) {
    __shared__ unsigned ldsLutSh[64];                     // 2 replicas of lut_exp<<7
    __shared__ float    ldsAcc[WAVES_PER_BLOCK * 256];    // 16 waves x 16x16 f32 tile

    const int tid      = threadIdx.x;
    const int lane     = tid & 31;
    const int wave     = tid >> 5;
    const int laneHalf = lane >> 4;       // 0: lanes 0-15, 1: lanes 16-31
    const int mrow     = lane & 15;       // A-matrix row (batch) held by this lane
    const int n0       = blockIdx.x * 16; // output-feature tile base

    if (tid < 32) {
        const unsigned v = ((unsigned)lut_exp[tid]) << 7;
        ldsLutSh[tid]      = v;   // replica 0: banks 0-31
        ldsLutSh[tid + 32] = v;   // replica 1: banks 32-63
    }
    __syncthreads();

    const unsigned lutBase = (unsigned)(lane & 1) << 5;   // odd lanes use upper replica

    const int kBegin = wave * K_CHUNK;
    const int kEnd   = kBegin + K_CHUNK;

    // Two independent accumulators: breaks the WMMA C-chain and doubles the
    // number of metadata loads in flight per iteration.
    v8f c0 = {0.f, 0.f, 0.f, 0.f, 0.f, 0.f, 0.f, 0.f};
    v8f c1 = {0.f, 0.f, 0.f, 0.f, 0.f, 0.f, 0.f, 0.f};

    const float* xrow = x + (size_t)mrow * IN_F;

    for (int k0 = kBegin; k0 < kEnd; k0 += 64) {
        c0 = df11_step(k0,      c0, xrow, exp_idx, sign_mant, ldsLutSh, lutBase, lane, laneHalf, n0);
        c1 = df11_step(k0 + 32, c1, xrow, exp_idx, sign_mant, ldsLutSh, lutBase, lane, laneHalf, n0);
    }

    // ---- deterministic cross-wave K reduction via LDS ----
#pragma unroll
    for (int r = 0; r < 8; ++r)
        ldsAcc[wave * 256 + lane * 8 + r] = c0[r] + c1[r];
    __syncthreads();

    if (tid < 32) {
        // C/D layout: VGPR r -> M = r + 8*laneHalf ; N = lane & 15
        const int n  = n0 + mrow;
        const float bv = bias[n];
#pragma unroll
        for (int r = 0; r < 8; ++r) {
            float s = 0.f;
#pragma unroll
            for (int w = 0; w < WAVES_PER_BLOCK; ++w)
                s += ldsAcc[w * 256 + lane * 8 + r];
            const int m = r + 8 * laneHalf;
            out[(size_t)m * OUT_F + n] = s + bv;
        }
    }
}

extern "C" void kernel_launch(void* const* d_in, const int* in_sizes, int n_in,
                              void* d_out, int out_size, void* d_ws, size_t ws_size,
                              hipStream_t stream) {
    (void)in_sizes; (void)n_in; (void)out_size; (void)d_ws; (void)ws_size;
    const float* x         = (const float*)d_in[0];
    const int*   exp_idx   = (const int*)d_in[1];
    const int*   sign_mant = (const int*)d_in[2];
    const int*   lut_exp   = (const int*)d_in[3];
    const float* bias      = (const float*)d_in[4];
    float*       out       = (float*)d_out;

    dim3 grid(OUT_F / 16);                // 256 N-tiles
    dim3 block(WAVES_PER_BLOCK * 32);     // 16 wave32 per block, K split across waves
    hipLaunchKernelGGL(df11_linear_wmma_kernel, grid, block, 0, stream,
                       x, exp_idx, sign_mant, lut_exp, bias, out);
}
